// Decoder1Cell_79800492359822
// MI455X (gfx1250) — compile-verified, matrix operand order checked by reading, not run on hardware
//
#include <hip/hip_runtime.h>
#include <hip/hip_bf16.h>
#include <math.h>

// ---------------------------------------------------------------------------
// CDNA5 WMMA (wave32): A/B are 16x32 / 32x16 bf16 (8 VGPRs = v16bf),
// C/D is 16x16 f32 (8 VGPRs = v8f).
// ---------------------------------------------------------------------------
typedef __attribute__((ext_vector_type(16))) __bf16 v16bf;
typedef __attribute__((ext_vector_type(8)))  __bf16 v8bf;
typedef __attribute__((ext_vector_type(8)))  float  v8f;

#define WMMA_BF16(a, b, c) \
    __builtin_amdgcn_wmma_f32_16x16x32_bf16(false, (a), false, (b), (short)0, (c), false, false)

// A-fragment from f32 source (activations): convert in-registers.
// ISA layout (CDNA5 ISA, 16-bit A 16x32): lane(0-15)=row, lanes 16-31 same
// rows / other K half. Lane holds K = hi*8 + [0..7] (elems 0..7) and
// K = 16 + hi*8 + [0..7] (elems 8..15).
__device__ __forceinline__ v16bf load_frag_f32(const float* __restrict__ rowp,
                                               int k0, int hi) {
    const float4* p0 = reinterpret_cast<const float4*>(rowp + k0 + hi * 8);
    const float4* p1 = reinterpret_cast<const float4*>(rowp + k0 + 16 + hi * 8);
    float4 a0 = p0[0], a1 = p0[1], b0 = p1[0], b1 = p1[1];
    v16bf f;
    f[0]  = (__bf16)a0.x; f[1]  = (__bf16)a0.y; f[2]  = (__bf16)a0.z; f[3]  = (__bf16)a0.w;
    f[4]  = (__bf16)a1.x; f[5]  = (__bf16)a1.y; f[6]  = (__bf16)a1.z; f[7]  = (__bf16)a1.w;
    f[8]  = (__bf16)b0.x; f[9]  = (__bf16)b0.y; f[10] = (__bf16)b0.z; f[11] = (__bf16)b0.w;
    f[12] = (__bf16)b1.x; f[13] = (__bf16)b1.y; f[14] = (__bf16)b1.z; f[15] = (__bf16)b1.w;
    return f;
}

// B-fragment from pre-packed bf16 weights: two 16B loads, no conversion.
__device__ __forceinline__ v16bf load_frag_bf16(const __bf16* __restrict__ rowp,
                                                int k0, int hi) {
    v8bf lo = *reinterpret_cast<const v8bf*>(rowp + k0 + hi * 8);
    v8bf hi8 = *reinterpret_cast<const v8bf*>(rowp + k0 + 16 + hi * 8);
    return __builtin_shufflevector(lo, hi8, 0, 1, 2, 3, 4, 5, 6, 7,
                                            8, 9, 10, 11, 12, 13, 14, 15);
}

// f32 -> bf16 packing (weights), 8 elements per thread.
__global__ void __launch_bounds__(256)
k_pack_bf16(const float* __restrict__ src, __bf16* __restrict__ dst, int n8) {
    int i = blockIdx.x * blockDim.x + threadIdx.x;
    if (i >= n8) return;
    const float4* s = reinterpret_cast<const float4*>(src) + 2 * (size_t)i;
    float4 a = s[0], b = s[1];
    v8bf o;
    o[0] = (__bf16)a.x; o[1] = (__bf16)a.y; o[2] = (__bf16)a.z; o[3] = (__bf16)a.w;
    o[4] = (__bf16)b.x; o[5] = (__bf16)b.y; o[6] = (__bf16)b.z; o[7] = (__bf16)b.w;
    reinterpret_cast<v8bf*>(dst)[i] = o;
}

// C = act(A(MxK,f32) @ W(NxK,bf16)^T + bias). M%32==0, N%64==0, K%32==0.
// Block = 8 waves covering 32x512 of C; each wave owns 2 M-tiles x 4 N-tiles
// (8 accumulators): one B-fragment feeds 2 WMMAs, one A-fragment feeds 4.
__global__ void __launch_bounds__(256)
k_gemm_bf16w(const float* __restrict__ A, const __bf16* __restrict__ W,
             const float* __restrict__ bias, float* __restrict__ C,
             int M, int N, int K, int act) {
    const int wave = threadIdx.x >> 5;
    const int lane = threadIdx.x & 31;
    const int row  = lane & 15;   // A row (M) for A-frag, W row (N) for B-frag
    const int hi   = lane >> 4;   // K half select
    const int m0 = blockIdx.y * 32;
    const int n0 = blockIdx.x * 512 + wave * 64;
    if (n0 >= N || m0 >= M) return;

    const float*  __restrict__ Arow0 = A + (size_t)(m0 + row) * K;
    const float*  __restrict__ Arow1 = Arow0 + (size_t)16 * K;
    const __bf16* __restrict__ W0    = W + (size_t)(n0 + row) * K;
    const size_t wstride = (size_t)16 * K;

    v8f acc[2][4] = {};
    for (int k0 = 0; k0 < K; k0 += 32) {
        v16bf a0 = load_frag_f32(Arow0, k0, hi);
        v16bf a1 = load_frag_f32(Arow1, k0, hi);
        v16bf b0 = load_frag_bf16(W0,               k0, hi);
        v16bf b1 = load_frag_bf16(W0 +     wstride, k0, hi);
        v16bf b2 = load_frag_bf16(W0 + 2 * wstride, k0, hi);
        v16bf b3 = load_frag_bf16(W0 + 3 * wstride, k0, hi);
        acc[0][0] = WMMA_BF16(a0, b0, acc[0][0]);
        acc[0][1] = WMMA_BF16(a0, b1, acc[0][1]);
        acc[0][2] = WMMA_BF16(a0, b2, acc[0][2]);
        acc[0][3] = WMMA_BF16(a0, b3, acc[0][3]);
        acc[1][0] = WMMA_BF16(a1, b0, acc[1][0]);
        acc[1][1] = WMMA_BF16(a1, b1, acc[1][1]);
        acc[1][2] = WMMA_BF16(a1, b2, acc[1][2]);
        acc[1][3] = WMMA_BF16(a1, b3, acc[1][3]);
    }

    // C/D layout: lane(0-15)->N=lane, elem i -> M = i + (lane>=16 ? 8 : 0).
    #pragma unroll
    for (int t = 0; t < 4; ++t) {
        const int n = n0 + 16 * t + row;
        const float bv = bias ? bias[n] : 0.0f;
        #pragma unroll
        for (int mi = 0; mi < 2; ++mi) {
            #pragma unroll
            for (int i = 0; i < 8; ++i) {
                float v = acc[mi][t][i] + bv;
                if (act) v = fmaxf(v, 0.0f);
                C[(size_t)(m0 + mi * 16 + hi * 8 + i) * N + n] = v;
            }
        }
    }
}

// GRU gate fusion: out = (1-z)*n + z*h (+ residual), optional 2nd (strided) copy.
__global__ void __launch_bounds__(256)
k_gru_combine(const float* __restrict__ gi, const float* __restrict__ gh,
              const float* __restrict__ hprev, const float* __restrict__ res,
              float* __restrict__ out1, float* __restrict__ out2, int out2_ld,
              int B, int H) {
    int idx = blockIdx.x * blockDim.x + threadIdx.x;
    if (idx >= B * H) return;
    int b = idx / H, j = idx - b * H;
    const float* gir = gi + (size_t)b * 3 * H;
    const float* ghr = gh + (size_t)b * 3 * H;
    float ir = gir[j], iz = gir[H + j], in_ = gir[2 * H + j];
    float hr = ghr[j], hz = ghr[H + j], hn  = ghr[2 * H + j];
    float r = 1.0f / (1.0f + __expf(-(ir + hr)));
    float z = 1.0f / (1.0f + __expf(-(iz + hz)));
    float n = tanhf(in_ + r * hn);
    float h = hprev[idx];
    float o = (1.0f - z) * n + z * h;
    if (res) o += res[idx];
    out1[idx] = o;
    if (out2) out2[(size_t)b * out2_ld + j] = o;
}

// Fused single pass over `memory` (256MB): ctx[b,c] = sum_t w[b,t]*mem[b,t,c]
// (into xdec[:,256:768]) and e[b,t] = dot(mem[b,t,:], q[b,:])/sqrt(C).
__global__ void __launch_bounds__(256)
k_ctx_scores(const float* __restrict__ memory, const float* __restrict__ w,
             const float* __restrict__ q, float* __restrict__ e,
             float* __restrict__ xdec) {
    constexpr int T = 512, C = 512, DR = 768;
    const int b    = blockIdx.x;
    const int wave = threadIdx.x >> 5;
    const int lane = threadIdx.x & 31;
    const int cb   = lane * 16;
    const float* __restrict__ mem = memory + (size_t)b * T * C;

    float qr[16], cacc[16];
    #pragma unroll
    for (int j = 0; j < 16; ++j) { qr[j] = q[(size_t)b * C + cb + j]; cacc[j] = 0.0f; }

    const float scale = 0.044194173824159216f;  // 1/sqrt(512)
    for (int t = wave; t < T; t += 8) {
        const float* rowp = mem + (size_t)t * C + cb;
        __builtin_prefetch(rowp + 8 * C, 0, 1);   // -> global_prefetch_b8
        const float wt = w[(size_t)b * T + t];
        float ep = 0.0f;
        #pragma unroll
        for (int v = 0; v < 4; ++v) {
            float4 m4 = reinterpret_cast<const float4*>(rowp)[v];
            ep += m4.x * qr[4 * v + 0] + m4.y * qr[4 * v + 1] +
                  m4.z * qr[4 * v + 2] + m4.w * qr[4 * v + 3];
            cacc[4 * v + 0] += wt * m4.x; cacc[4 * v + 1] += wt * m4.y;
            cacc[4 * v + 2] += wt * m4.z; cacc[4 * v + 3] += wt * m4.w;
        }
        #pragma unroll
        for (int off = 16; off >= 1; off >>= 1) ep += __shfl_xor(ep, off, 32);
        if (lane == 0) e[(size_t)b * T + t] = ep * scale;
    }

    __shared__ float ctx_s[C];
    ctx_s[threadIdx.x] = 0.0f; ctx_s[threadIdx.x + 256] = 0.0f;
    __syncthreads();
    #pragma unroll
    for (int j = 0; j < 16; ++j) atomicAdd(&ctx_s[cb + j], cacc[j]);
    __syncthreads();
    float* ctx_out = xdec + (size_t)b * DR + 256;
    ctx_out[threadIdx.x]       = ctx_s[threadIdx.x];
    ctx_out[threadIdx.x + 256] = ctx_s[threadIdx.x + 256];
}

// softmax(e) * 0.5*(w + shift(w)), renormalized (+1e-8). One block per batch row.
__global__ void __launch_bounds__(256)
k_softmax_markov(const float* __restrict__ e, const float* __restrict__ w,
                 float* __restrict__ w_new) {
    constexpr int T = 512;
    const int b = blockIdx.x, tid = threadIdx.x;
    __shared__ float red[256];
    const float e0 = e[(size_t)b * T + tid];
    const float e1 = e[(size_t)b * T + tid + 256];

    red[tid] = fmaxf(e0, e1); __syncthreads();
    for (int s = 128; s > 0; s >>= 1) { if (tid < s) red[tid] = fmaxf(red[tid], red[tid + s]); __syncthreads(); }
    const float m = red[0]; __syncthreads();

    const float x0 = __expf(e0 - m), x1 = __expf(e1 - m);
    red[tid] = x0 + x1; __syncthreads();
    for (int s = 128; s > 0; s >>= 1) { if (tid < s) red[tid] += red[tid + s]; __syncthreads(); }
    const float denom = red[0]; __syncthreads();

    const float a0 = x0 / denom, a1 = x1 / denom;
    const float w0 = w[(size_t)b * T + tid], w1 = w[(size_t)b * T + tid + 256];
    const float wm0 = (tid == 0) ? 0.0f : w[(size_t)b * T + tid - 1];
    const float wm1 = w[(size_t)b * T + tid + 255];
    const float u0 = a0 * 0.5f * (w0 + wm0);
    const float u1 = a1 * 0.5f * (w1 + wm1);

    red[tid] = u0 + u1; __syncthreads();
    for (int s = 128; s > 0; s >>= 1) { if (tid < s) red[tid] += red[tid + s]; __syncthreads(); }
    const float inv = 1.0f / (red[0] + 1e-8f);
    w_new[(size_t)b * T + tid]       = u0 * inv;
    w_new[(size_t)b * T + tid + 256] = u1 * inv;
}

// ---------------------------------------------------------------------------
extern "C" void kernel_launch(void* const* d_in, const int* in_sizes, int n_in,
                              void* d_out, int out_size, void* d_ws, size_t ws_size,
                              hipStream_t stream) {
    constexpr int B = 256, T = 512, Cc = 512, PRE = 128, ATT = 256, RNN = 768, MEL = 160;

    const float* x       = (const float*)d_in[0];
    const float* w       = (const float*)d_in[1];
    const float* h_att   = (const float*)d_in[2];
    const float* h_dec0  = (const float*)d_in[3];
    const float* h_dec1  = (const float*)d_in[4];
    const float* memory  = (const float*)d_in[5];
    const float* pre_w1  = (const float*)d_in[6];
    const float* pre_b1  = (const float*)d_in[7];
    const float* pre_w2  = (const float*)d_in[8];
    const float* pre_b2  = (const float*)d_in[9];
    const float* att_wi  = (const float*)d_in[10];
    const float* att_wh  = (const float*)d_in[11];
    const float* att_bi  = (const float*)d_in[12];
    const float* att_bh  = (const float*)d_in[13];
    const float* q_w     = (const float*)d_in[14];
    const float* q_b     = (const float*)d_in[15];
    const float* dec0_wi = (const float*)d_in[16];
    const float* dec0_wh = (const float*)d_in[17];
    const float* dec0_bi = (const float*)d_in[18];
    const float* dec0_bh = (const float*)d_in[19];
    const float* dec1_wi = (const float*)d_in[20];
    const float* dec1_wh = (const float*)d_in[21];
    const float* dec1_bi = (const float*)d_in[22];
    const float* dec1_bh = (const float*)d_in[23];

    // Output layout: h1 | w_new | h_att_new | h0 | h1
    float* out = (float*)d_out;
    float* o_h1   = out;
    float* o_wnew = out + (size_t)B * RNN;
    float* o_hatt = o_wnew + (size_t)B * T;
    float* o_h0   = o_hatt + (size_t)B * ATT;
    float* o_h1b  = o_h0 + (size_t)B * RNN;

    // ---- Workspace: f32 scratch region, then bf16 weight region -----------
    float* ws   = (float*)d_ws;
    float* pre1 = ws;                               // B*128
    float* pre2 = pre1 + (size_t)B * PRE;           // B*128
    float* gi   = pre2 + (size_t)B * PRE;           // B*2304
    float* gh   = gi   + (size_t)B * 3 * RNN;       // B*2304
    float* qbuf = gh   + (size_t)B * 3 * RNN;       // B*512
    float* ebuf = qbuf + (size_t)B * Cc;            // B*512
    float* xdec = ebuf + (size_t)B * T;             // B*768
    __bf16* wb  = (__bf16*)(xdec + (size_t)B * RNN);

    __bf16* wb_pre1 = wb;                                    // 128*160
    __bf16* wb_pre2 = wb_pre1 + (size_t)PRE * MEL;           // 128*128
    __bf16* wb_awi  = wb_pre2 + (size_t)PRE * PRE;           // 768*128
    __bf16* wb_awh  = wb_awi  + (size_t)3 * ATT * PRE;       // 768*256
    __bf16* wb_qw   = wb_awh  + (size_t)3 * ATT * ATT;       // 512*256
    __bf16* wb_d0wi = wb_qw   + (size_t)Cc * ATT;            // 2304*768
    __bf16* wb_d0wh = wb_d0wi + (size_t)3 * RNN * RNN;
    __bf16* wb_d1wi = wb_d0wh + (size_t)3 * RNN * RNN;
    __bf16* wb_d1wh = wb_d1wi + (size_t)3 * RNN * RNN;

    auto pack = [&](const float* src, __bf16* dst, size_t n) {
        int n8 = (int)(n / 8);
        k_pack_bf16<<<(n8 + 255) / 256, 256, 0, stream>>>(src, dst, n8);
    };
    auto gemm = [&](const float* A, const __bf16* W, const float* bias, float* C,
                    int M, int N, int K, int act) {
        dim3 grid((N + 511) / 512, M / 32);
        k_gemm_bf16w<<<grid, 256, 0, stream>>>(A, W, bias, C, M, N, K, act);
    };

    // 0) Pack all weights to bf16 once (independent of everything else).
    pack(pre_w1,  wb_pre1, (size_t)PRE * MEL);
    pack(pre_w2,  wb_pre2, (size_t)PRE * PRE);
    pack(att_wi,  wb_awi,  (size_t)3 * ATT * PRE);
    pack(att_wh,  wb_awh,  (size_t)3 * ATT * ATT);
    pack(q_w,     wb_qw,   (size_t)Cc * ATT);
    pack(dec0_wi, wb_d0wi, (size_t)3 * RNN * RNN);
    pack(dec0_wh, wb_d0wh, (size_t)3 * RNN * RNN);
    pack(dec1_wi, wb_d1wi, (size_t)3 * RNN * RNN);
    pack(dec1_wh, wb_d1wh, (size_t)3 * RNN * RNN);

    // 1) PreNet
    gemm(x,    wb_pre1, pre_b1, pre1, B, PRE, MEL, 1);
    gemm(pre1, wb_pre2, pre_b2, pre2, B, PRE, PRE, 1);

    // 2) Attention GRU: gates, then fuse; h_att_new -> d_out + xdec[:, :256]
    gemm(pre2,  wb_awi, att_bi, gi, B, 3 * ATT, PRE, 0);
    gemm(h_att, wb_awh, att_bh, gh, B, 3 * ATT, ATT, 0);
    k_gru_combine<<<(B * ATT + 255) / 256, 256, 0, stream>>>(
        gi, gh, h_att, nullptr, o_hatt, xdec, RNN, B, ATT);

    // 3) q projection
    gemm(o_hatt, wb_qw, q_b, qbuf, B, Cc, ATT, 0);

    // 4) Single fused streaming pass over `memory`: ctx (into xdec) + scores e
    k_ctx_scores<<<B, 256, 0, stream>>>(memory, w, qbuf, ebuf, xdec);

    // 5) softmax * Markov prior -> w_new
    k_softmax_markov<<<B, 256, 0, stream>>>(ebuf, w, o_wnew);

    // 6) ResGRUCell 0: h0 = GRU(xdec, h_dec0) + xdec
    gemm(xdec,   wb_d0wi, dec0_bi, gi, B, 3 * RNN, RNN, 0);
    gemm(h_dec0, wb_d0wh, dec0_bh, gh, B, 3 * RNN, RNN, 0);
    k_gru_combine<<<(B * RNN + 255) / 256, 256, 0, stream>>>(
        gi, gh, h_dec0, xdec, o_h0, nullptr, 0, B, RNN);

    // 7) ResGRUCell 1: h1 = GRU(h0, h_dec1) + h0  (both h1 slots)
    gemm(o_h0,   wb_d1wi, dec1_bi, gi, B, 3 * RNN, RNN, 0);
    gemm(h_dec1, wb_d1wh, dec1_bh, gh, B, 3 * RNN, RNN, 0);
    k_gru_combine<<<(B * RNN + 255) / 256, 256, 0, stream>>>(
        gi, gh, h_dec1, o_h0, o_h1, o_h1b, RNN, B, RNN);
}
